// GNN_81784767250667
// MI455X (gfx1250) — matrix operand order repaired and checked by reading, not compiled, not run on hardware
//
#include <hip/hip_runtime.h>
#include <math.h>

// ---------------------------------------------------------------------------
// GATv2 x2 + BN + PReLU for MI455X (gfx1250, wave32, WMMA).
// Fixed problem shape from reference: N=50000, E=800000, H=4, D=128.
// ---------------------------------------------------------------------------

#define DH   128          // hidden dim
#define HH   4            // heads
#define HD   (HH * DH)    // 512
#define BN_EPS 1e-5f
#define NEG_SLOPE 0.2f

typedef __attribute__((ext_vector_type(16))) _Float16 v16h;
typedef __attribute__((ext_vector_type(8)))  float    v8f;

// ---- float atomic max via sign-aware integer atomics ----------------------
__device__ __forceinline__ void atomicMaxF(float* addr, float val) {
  if (val >= 0.0f) atomicMax((int*)addr, __float_as_int(val));
  else             atomicMin((unsigned int*)addr, __float_as_uint(val));
}

// ---------------------------------------------------------------------------
// C[M,P] = A[M,128] * W[128,P] + bias[P]   (A fp32 -> f16, f32 accumulate)
// P is compile-time so all W / C offsets fold into immediate VMEM offsets.
// Block = 256 thr = 8 waves. Block covers one 16-row strip of A (staged in
// LDS as f16) and 8 consecutive 16-col tiles of W. Fully unrolled K: four
// back-to-back v_wmma_f32_16x16x32_f16 per wave. No divergence -> EXEC all
// ones at every WMMA as the ISA requires.
// ---------------------------------------------------------------------------
template <int P>
__global__ void __launch_bounds__(256)
gemm_wmma(const float* __restrict__ A, const float* __restrict__ W,
          const float* __restrict__ bias, float* __restrict__ C) {
  constexpr int GROUPS = P / 128;             // 16-col tiles per block = 8
  __shared__ _Float16 As[16][136];            // pad 128->136 to spread banks

  const int tileM = blockIdx.x / GROUPS;
  const int tg    = blockIdx.x % GROUPS;
  const int wave  = threadIdx.x >> 5;
  const int lane  = threadIdx.x & 31;

  // Stage 16x128 fp32 A-strip into LDS as f16, coalesced.
  const float* Arow = A + (size_t)tileM * 16 * DH;
  for (int t = threadIdx.x; t < 16 * DH; t += 256) {
    const int r = t >> 7, k = t & 127;
    As[r][k] = (_Float16)Arow[r * DH + k];
  }
  __syncthreads();

  const int tileN = tg * 8 + wave;
  const int rl    = lane & 15;
  const int half  = lane >> 4;
  const int col   = tileN * 16 + rl;
  const float* Wc = W + col;                  // immediate offsets: k*P*4 B

  v8f c = {};
#pragma unroll
  for (int k0 = 0; k0 < DH; k0 += 32) {
    v16h a, b;
#pragma unroll
    for (int i = 0; i < 16; ++i) {
      // A 16x32 f16 layout: lanes 0-15 hold K {0..7,16..23}, lanes 16-31 {8..15,24..31}
      const int ka = k0 + ((i < 8) ? i : i + 8) + half * 8;
      a[i] = As[rl][ka];
      // B 32x16 f16 layout: lanes 0-15 hold K {0..7,16..23}, lanes 16-31 {8..15,24..31}
      const int kb = k0 + ((i < 8) ? i : i + 8) + half * 8;
      b[i] = (_Float16)Wc[kb * P];
    }
    c = __builtin_amdgcn_wmma_f32_16x16x32_f16(false, a, false, b,
                                               (short)0, c, false, false);
  }

  // C/D layout: VGPR r -> (M = r + (half?8:0), N = lane&15)
  const float bv = bias[col];
  float* Cp = C + (size_t)(tileM * 16 + (half ? 8 : 0)) * P + col;
#pragma unroll
  for (int r = 0; r < 8; ++r)
    Cp[r * P] = c[r] + bv;
}

// ---------------------------------------------------------------------------
// init lmax=-inf, denom=0, bn sums=0
// ---------------------------------------------------------------------------
__global__ void init_buffers(float* __restrict__ lmax, float* __restrict__ denom,
                             float* __restrict__ bnsum, float* __restrict__ bnsumsq,
                             int NH) {
  const int i = blockIdx.x * blockDim.x + threadIdx.x;
  if (i < NH) { lmax[i] = -__builtin_inff(); denom[i] = 0.0f; }
  if (i < DH) { bnsum[i] = 0.0f; bnsumsq[i] = 0.0f; }
}

// ---------------------------------------------------------------------------
// Pass 1: logits[e,h] = sum_d att[h,d]*leaky_relu(xl[src,h,d]+xr[dst,h,d])
//         + atomic segment-max into lmax[dst,h].  One wave32 per edge;
//         float4 loads -> global_load_b128, wave32 shuffle reduction.
// ---------------------------------------------------------------------------
__global__ void __launch_bounds__(256)
edge_logits(const float* __restrict__ xl, const float* __restrict__ xr,
            const int* __restrict__ src, const int* __restrict__ dst,
            const float* __restrict__ att, float* __restrict__ logits,
            float* __restrict__ lmax, int E) {
  const int e    = (int)((blockIdx.x * blockDim.x + threadIdx.x) >> 5);
  const int lane = threadIdx.x & 31;
  if (e >= E) return;
  const int s  = src[e];
  const int dn = dst[e];
  const float4* xls = (const float4*)(xl + (size_t)s  * HD);
  const float4* xrd = (const float4*)(xr + (size_t)dn * HD);
  const float4* at4 = (const float4*)att;
#pragma unroll
  for (int h = 0; h < HH; ++h) {
    const int idx4 = h * (DH / 4) + lane;
    const float4 va = xls[idx4];
    const float4 vb = xrd[idx4];
    const float4 aa = at4[idx4];
    float m, p = 0.0f;
    m = va.x + vb.x; p += ((m >= 0.0f) ? m : NEG_SLOPE * m) * aa.x;
    m = va.y + vb.y; p += ((m >= 0.0f) ? m : NEG_SLOPE * m) * aa.y;
    m = va.z + vb.z; p += ((m >= 0.0f) ? m : NEG_SLOPE * m) * aa.z;
    m = va.w + vb.w; p += ((m >= 0.0f) ? m : NEG_SLOPE * m) * aa.w;
    for (int off = 16; off > 0; off >>= 1) p += __shfl_down(p, off, 32);
    if (lane == 0) {
      logits[(size_t)e * HH + h] = p;
      atomicMaxF(&lmax[dn * HH + h], p);
    }
  }
}

// ---------------------------------------------------------------------------
// Pass 2: denom[dst,h] += exp(logit - lmax[dst,h]).  Thread per (edge,head).
// ---------------------------------------------------------------------------
__global__ void __launch_bounds__(256)
edge_exp(const float* __restrict__ logits, const float* __restrict__ lmax,
         const int* __restrict__ dst, float* __restrict__ denom, int E) {
  const int idx = blockIdx.x * blockDim.x + threadIdx.x;
  if (idx >= E * HH) return;
  const int e = idx >> 2, h = idx & 3;
  const int dn = dst[e];
  const float ex = __expf(logits[idx] - lmax[dn * HH + h]);
  atomicAdd(&denom[dn * HH + h], ex);
}

// ---------------------------------------------------------------------------
// Pass 3: accum[dst,d] += (1/H) * sum_h alpha[e,h] * xl[src,h,d]
//         One wave32 per edge; lane handles 4 channels (float4 gather).
// ---------------------------------------------------------------------------
__global__ void __launch_bounds__(256)
edge_scatter(const float* __restrict__ xl, const float* __restrict__ logits,
             const float* __restrict__ lmax, const float* __restrict__ denom,
             const int* __restrict__ src, const int* __restrict__ dst,
             float* __restrict__ accum, int E) {
  const int e    = (int)((blockIdx.x * blockDim.x + threadIdx.x) >> 5);
  const int lane = threadIdx.x & 31;
  if (e >= E) return;
  const int s  = src[e];
  const int dn = dst[e];
  float al[HH];
#pragma unroll
  for (int h = 0; h < HH; ++h) {
    const float ex = __expf(logits[(size_t)e * HH + h] - lmax[dn * HH + h]);
    al[h] = ex / fmaxf(denom[dn * HH + h], 1e-16f) * (1.0f / HH);
  }
  const float4* xls = (const float4*)(xl + (size_t)s * HD);
  float ax = 0.f, ay = 0.f, az = 0.f, aw = 0.f;
#pragma unroll
  for (int h = 0; h < HH; ++h) {
    const float4 v = xls[h * (DH / 4) + lane];
    ax += al[h] * v.x; ay += al[h] * v.y; az += al[h] * v.z; aw += al[h] * v.w;
  }
  float* o = accum + (size_t)dn * DH + lane * 4;
  atomicAdd(o + 0, ax); atomicAdd(o + 1, ay);
  atomicAdd(o + 2, az); atomicAdd(o + 3, aw);
}

// ---------------------------------------------------------------------------
// BN stats: per-channel sum / sumsq. Block = 128 threads (one per channel),
// grid-stride over rows, single atomic per (block,channel).
// ---------------------------------------------------------------------------
__global__ void __launch_bounds__(128)
bn_stats(const float* __restrict__ X, float* __restrict__ sum,
         float* __restrict__ sumsq, int N) {
  const int c = threadIdx.x;
  float s = 0.0f, q = 0.0f;
  for (int r = blockIdx.x; r < N; r += gridDim.x) {
    const float v = X[(size_t)r * DH + c];
    s += v; q += v * v;
  }
  atomicAdd(&sum[c], s);
  atomicAdd(&sumsq[c], q);
}

// ---------------------------------------------------------------------------
// BN apply + PReLU; writes the layer output into xnext (feeds layer 2) and
// into the [N,256] concatenated final output at column offset colOff.
// ---------------------------------------------------------------------------
__global__ void __launch_bounds__(256)
bn_apply(const float* __restrict__ X, const float* __restrict__ sum,
         const float* __restrict__ sumsq, const float* __restrict__ g,
         const float* __restrict__ beta, const float* __restrict__ aprm,
         float* __restrict__ xnext, float* __restrict__ out,
         int N, int colOff) {
  const int idx = blockIdx.x * blockDim.x + threadIdx.x;
  if (idx >= N * DH) return;
  const int r = idx >> 7, c = idx & 127;
  const float invN = 1.0f / (float)N;
  const float mu   = sum[c] * invN;
  const float var  = sumsq[c] * invN - mu * mu;
  const float xh   = (X[idx] - mu) * rsqrtf(var + BN_EPS) * g[c] + beta[c];
  const float y    = (xh >= 0.0f) ? xh : aprm[c] * xh;
  xnext[idx] = y;
  out[(size_t)r * (2 * DH) + colOff + c] = y;
}

// ---------------------------------------------------------------------------
extern "C" void kernel_launch(void* const* d_in, const int* in_sizes, int n_in,
                              void* d_out, int out_size, void* d_ws, size_t ws_size,
                              hipStream_t stream) {
  const float* x  = (const float*)d_in[0];
  const int*   ei = (const int*)d_in[1];
  const int N = in_sizes[0] / DH;      // 50000
  const int E = in_sizes[1] / 2;       // 800000
  const int* src = ei;
  const int* dst = ei + E;

  // Workspace layout (floats), ~270 MB total.
  float* ws      = (float*)d_ws;
  const size_t nHD = (size_t)N * HD;
  float* xl      = ws;                          // N*512
  float* xr      = xl + nHD;                    // N*512
  float* accum   = xr + nHD;                    // N*128 (residual + scatter)
  float* x1      = accum + (size_t)N * DH;      // N*128 (layer output)
  float* logits  = x1 + (size_t)N * DH;         // E*4
  float* lmax    = logits + (size_t)E * HH;     // N*4
  float* denom   = lmax + (size_t)N * HH;       // N*4
  float* bnsum   = denom + (size_t)N * HH;      // 128
  float* bnsumsq = bnsum + DH;                  // 128
  float* outp    = (float*)d_out;

  const int tilesM = N / 16;                    // 3125
  const int NH     = N * HH;

  for (int layer = 0; layer < 2; ++layer) {
    const float* xin  = layer ? x1 : x;
    const int    b    = 2 + 10 * layer;
    const float* Wl   = (const float*)d_in[b + 0];
    const float* bl   = (const float*)d_in[b + 1];
    const float* Wr   = (const float*)d_in[b + 2];
    const float* br   = (const float*)d_in[b + 3];
    const float* att  = (const float*)d_in[b + 4];
    const float* Wres = (const float*)d_in[b + 5];
    const float* bias = (const float*)d_in[b + 6];
    const float* g    = (const float*)d_in[b + 7];
    const float* beta = (const float*)d_in[b + 8];
    const float* aprm = (const float*)d_in[b + 9];

    // xl = x@Wl+bl, xr = x@Wr+br  (P=512)
    gemm_wmma<HD><<<tilesM * (HD / 128), 256, 0, stream>>>(xin, Wl, bl, xl);
    gemm_wmma<HD><<<tilesM * (HD / 128), 256, 0, stream>>>(xin, Wr, br, xr);
    // accum = x@Wres + bias  (P=128) -- scatter target
    gemm_wmma<DH><<<tilesM, 256, 0, stream>>>(xin, Wres, bias, accum);

    init_buffers<<<(NH + 255) / 256, 256, 0, stream>>>(lmax, denom, bnsum, bnsumsq, NH);
    edge_logits<<<(E + 7) / 8, 256, 0, stream>>>(xl, xr, src, dst, att, logits, lmax, E);
    edge_exp<<<(E * HH + 255) / 256, 256, 0, stream>>>(logits, lmax, dst, denom, E);
    edge_scatter<<<(E + 7) / 8, 256, 0, stream>>>(xl, logits, lmax, denom, src, dst, accum, E);

    bn_stats<<<512, 128, 0, stream>>>(accum, bnsum, bnsumsq, N);
    bn_apply<<<(N * DH + 255) / 256, 256, 0, stream>>>(accum, bnsum, bnsumsq,
                                                       g, beta, aprm, x1, outp,
                                                       N, layer * DH);
  }
}
